// DynamicDepthwiseConv_47081431499042
// MI455X (gfx1250) — matrix operand order repaired and checked by reading, not compiled, not run on hardware
//
#include <hip/hip_runtime.h>
#include <math.h>

// Problem constants (match reference)
#define B_    16
#define C_    64
#define H_    256
#define W_    256
#define K_    7
#define HID_  16
#define COUT_ (C_ * K_ * K_)   // 3136
#define NCHAN (B_ * C_)        // 1024
#define PLANE (H_ * W_)        // 65536

// Conv-kernel tiling
#define PROWS 70               // 64 + 6 halo
#define PSTR  72               // LDS row stride (f16 elems), keeps 4B alignment, spreads banks
#define KTOT  154              // 7 dy * 22 j
// padded to 5 chunks of K=32 per WMMA

typedef __attribute__((ext_vector_type(16))) _Float16     v16h;
typedef __attribute__((ext_vector_type(8)))  float        v8f;
typedef __attribute__((ext_vector_type(8)))  unsigned int v8u;

__device__ __forceinline__ int reflect256(int s) {
    s = s < 0 ? -s : s;
    return s > 255 ? 510 - s : s;
}

// ---------------- Kernel 1: global average pool (one block per (b,c) plane) ----
__global__ __launch_bounds__(256) void gap_kernel(const float* __restrict__ x,
                                                  float* __restrict__ pooled) {
    const int chan = blockIdx.x;
    const int tid  = threadIdx.x;
    const float4* p = (const float4*)(x + (size_t)chan * PLANE);
    float s = 0.f;
    for (int i = tid; i < PLANE / 4; i += 256) {
        float4 v = p[i];
        s += v.x + v.y + v.z + v.w;
    }
    __shared__ float red[256];
    red[tid] = s;
    __syncthreads();
    for (int off = 128; off > 0; off >>= 1) {
        if (tid < off) red[tid] += red[tid + off];
        __syncthreads();
    }
    if (tid == 0) pooled[chan] = red[0] * (1.f / (float)PLANE);
}

// ---------------- Kernel 2a: hdn = gelu(pooled @ w1 + b1), 16x16 --------------
__global__ __launch_bounds__(256) void mlp1_kernel(const float* __restrict__ pooled,
                                                   const float* __restrict__ w1,
                                                   const float* __restrict__ b1,
                                                   float* __restrict__ hdn) {
    const int t = threadIdx.x;      // 256 = 16 b * 16 h
    const int b = t >> 4;
    const int h = t & 15;
    float s = b1[h];
    for (int k = 0; k < C_; ++k)
        s += pooled[b * C_ + k] * w1[k * HID_ + h];
    // exact GELU: x * 0.5 * (1 + erf(x / sqrt(2)))
    hdn[t] = 0.5f * s * (1.f + erff(s * 0.70710678118654752f));
}

// ---------------- Kernel 2b: kw = sigmoid(hdn @ w2 + b2), 16 x 3136 -----------
__global__ __launch_bounds__(256) void mlp2_kernel(const float* __restrict__ hdn,
                                                   const float* __restrict__ w2,
                                                   const float* __restrict__ b2,
                                                   float* __restrict__ kw) {
    const int idx = blockIdx.x * 256 + threadIdx.x;
    if (idx >= B_ * COUT_) return;
    const int b = idx / COUT_;
    const int o = idx - b * COUT_;
    float s = b2[o];
#pragma unroll
    for (int k = 0; k < HID_; ++k)
        s += hdn[b * HID_ + k] * w2[k * COUT_ + o];
    // flat layout: kw[(b*64 + c)*49 + q]  (matches reshape(b*c,1,7,7))
    kw[idx] = 1.f / (1.f + expf(-s));
}

// ---------------- Kernel 3: dynamic depthwise 7x7 conv via WMMA ---------------
// One block = one channel, one 64x64 output region. 8 waves, 2 tiles each.
// D(16x16) = sum over 5 chunks of A(16x32 f16) x B(32x16 f16), fp32 accum.
// K index t in [0,160): dy = t/22, j = t%22. A[r,t] = patch[r+dy][tileX+j],
// B[t,n] = w[dy][j-n] if 0<=j-n<7 else 0 (Toeplitz of the dynamic kernel).
__global__ __launch_bounds__(256) void dyconv_wmma_kernel(const float* __restrict__ x,
                                                          const float* __restrict__ kw,
                                                          float* __restrict__ out) {
    __shared__ _Float16 patch[PROWS * PSTR];
    __shared__ float    kwS[K_ * K_];

    const int bx   = blockIdx.x;
    const int chan = bx >> 4;
    const int reg  = bx & 15;
    const int row0 = (reg >> 2) * 64;
    const int col0 = (reg & 3) * 64;
    const int tid  = threadIdx.x;
    const size_t chanOff = (size_t)chan * PLANE;

    if (tid < K_ * K_) kwS[tid] = kw[chan * (K_ * K_) + tid];

    // Stage reflect-padded 70x70 patch (f32 -> f16) into LDS.
    for (int i = tid; i < PROWS * PROWS; i += 256) {
        int pr = i / PROWS;
        int pc = i - pr * PROWS;
        int sr = reflect256(row0 + pr - 3);
        int sc = reflect256(col0 + pc - 3);
        patch[pr * PSTR + pc] = (_Float16)x[chanOff + (size_t)sr * W_ + sc];
    }
    __syncthreads();

    const int lane = tid & 31;
    const int wv   = tid >> 5;
    const int g    = lane >> 4;   // half-wave group
    const int r    = lane & 15;   // A-matrix row (M) for this lane
    const int n    = r;           // B/D column (N) for this lane

    // Per-lane A byte offsets relative to tile origin in the patch.
    // 16-bit A 16x32 layout: lanes 0-15 hold K {0..7, 16..23}; lanes 16-31 +8.
    int offA[5][8];
#pragma unroll
    for (int kc = 0; kc < 5; ++kc) {
#pragma unroll
        for (int v = 0; v < 8; ++v) {
            int klo = (v < 4 ? 2 * v : 16 + 2 * (v - 4)) + 8 * g;
            int t   = kc * 32 + klo;
            int dy  = t / 22;
            int j   = t - dy * 22;
            offA[kc][v] = (t < KTOT) ? ((r + dy) * PSTR + j) * 2 : 0;
        }
    }

    // Build per-channel B operand (held in registers across all tiles).
    // 16-bit B 32x16 layout (matches A): lanes 0-15: N=lane, K {0..7, 16..23}; lanes 16-31: N=lane-16, K +8.
    v16h bmat[5];
#pragma unroll
    for (int kc = 0; kc < 5; ++kc) {
        v8u bu;
#pragma unroll
        for (int v = 0; v < 8; ++v) {
            unsigned pk = 0;
#pragma unroll
            for (int half = 0; half < 2; ++half) {
                int t = kc * 32 + (v < 4 ? 2 * v : 16 + 2 * (v - 4)) + 8 * g + half;
                unsigned hv = 0;
                if (t < KTOT) {
                    int dy = t / 22;
                    int j  = t - dy * 22;
                    int dx = j - n;
                    if (dx >= 0 && dx < K_) {
                        _Float16 w = (_Float16)kwS[dy * K_ + dx];
                        hv = (unsigned)__builtin_bit_cast(unsigned short, w);
                    }
                }
                pk |= hv << (16 * half);
            }
            bu[v] = pk;
        }
        bmat[kc] = __builtin_bit_cast(v16h, bu);
    }

    // Each wave computes two 16x16 output tiles of the 64x64 region.
#pragma unroll
    for (int it = 0; it < 2; ++it) {
        const int ti = 2 * wv + it;
        const int ty = ti >> 2;
        const int tx = ti & 3;
        const char* pbase = (const char*)patch + ((ty * 16) * PSTR + tx * 16) * 2;

        v8f acc = {};
#pragma unroll
        for (int kc = 0; kc < 5; ++kc) {
            v8u au;
#pragma unroll
            for (int v = 0; v < 8; ++v)
                au[v] = *(const unsigned int*)(pbase + offA[kc][v]);
            v16h a = __builtin_bit_cast(v16h, au);
            acc = __builtin_amdgcn_wmma_f32_16x16x32_f16(
                /*neg_a=*/false, a, /*neg_b=*/false, bmat[kc],
                /*c_mod=*/(short)0, acc, /*reuse_a=*/false, /*reuse_b=*/false);
        }

        // f32 C/D layout: VGPR v -> M = v + 8*g, N = lane&15. Coalesced stores.
        float* o = out + chanOff + (size_t)(row0 + ty * 16 + 8 * g) * W_ + (col0 + tx * 16 + n);
#pragma unroll
        for (int v = 0; v < 8; ++v)
            o[(size_t)v * W_] = acc[v];
    }
}

// ---------------- Host-side launch --------------------------------------------
extern "C" void kernel_launch(void* const* d_in, const int* in_sizes, int n_in,
                              void* d_out, int out_size, void* d_ws, size_t ws_size,
                              hipStream_t stream) {
    const float* x  = (const float*)d_in[0];
    const float* w1 = (const float*)d_in[1];
    const float* b1 = (const float*)d_in[2];
    const float* w2 = (const float*)d_in[3];
    const float* b2 = (const float*)d_in[4];
    float* out = (float*)d_out;

    // Workspace layout (floats): pooled[1024] | hdn[256] | kw[50176] ~ 206 KB
    float* pooled = (float*)d_ws;
    float* hdn    = pooled + NCHAN;
    float* kwbuf  = hdn + B_ * HID_;

    gap_kernel<<<NCHAN, 256, 0, stream>>>(x, pooled);
    mlp1_kernel<<<1, 256, 0, stream>>>(pooled, w1, b1, hdn);
    mlp2_kernel<<<(B_ * COUT_ + 255) / 256, 256, 0, stream>>>(hdn, w2, b2, kwbuf);
    dyconv_wmma_kernel<<<NCHAN * 16, 256, 0, stream>>>(x, kwbuf, out);
}
